// RNN_ATT_NOISE_TM1_22651657519484
// MI455X (gfx1250) — matrix operand order repaired and checked by reading, not compiled, not run on hardware
//
#include <hip/hip_runtime.h>
#include <stdint.h>

#define B_SZ 256
#define C_SZ 8
#define T_SZ 1024
#define H_SZ 512
#define WB_INCf 0.05f

// output section base offsets (in floats), concatenated return order:
// keys (B,2,T), prs (B,4,T), hs (B,H,T), tm_modified (B,2,T)
#define PK 0
#define PP (B_SZ * 2 * T_SZ)            /* 524288    */
#define PH (PP + B_SZ * 4 * T_SZ)       /* 1572864   */
#define PT (PH + B_SZ * H_SZ * T_SZ)    /* 135790592 */

#define KC 256                           /* K-range cached in LDS (bf16)      */

typedef __attribute__((ext_vector_type(16))) __bf16 v16bf;
typedef __attribute__((ext_vector_type(8)))  __bf16 bf16x8;
typedef __attribute__((ext_vector_type(8)))  float  v8f;

__device__ __forceinline__ unsigned short f32_to_bf16(float x) {
    unsigned int u = __float_as_uint(x);
    u += 0x7FFFu + ((u >> 16) & 1u);          // round-to-nearest-even
    return (unsigned short)(u >> 16);
}

__device__ __forceinline__ float lik(float tm, float a) {
    if (tm == 0.f) return 0.f;
    float r = (tm - a) / (0.15f * a);
    return r * r;
}

__device__ __forceinline__ float fast_tanh(float x) {
    x = fminf(fmaxf(x, -20.f), 20.f);
    float e = __expf(2.f * x);
    return (e - 1.f) / (e + 1.f);
}

// ---------------------------------------------------------------------------
// Prep: w_rec f32 -> bf16 (RNE), idx_tm1 = argmax_t tm1[b,t]
// ---------------------------------------------------------------------------
__global__ __launch_bounds__(256) void rnn_prep(
    const float* __restrict__ w_rec, unsigned short* __restrict__ wrec_bf,
    const float* __restrict__ inputs, int* __restrict__ idx_tm1)
{
    int tid = blockIdx.x * 256 + threadIdx.x;
    for (int i = tid; i < H_SZ * H_SZ; i += gridDim.x * 256)
        wrec_bf[i] = f32_to_bf16(w_rec[i]);
    if (blockIdx.x == 0) {
        int b = threadIdx.x;                      // 256 threads, one per batch row
        const float* tm1 = inputs + b * C_SZ * T_SZ + 6 * T_SZ;
        float best = tm1[0]; int bi = 0;
        for (int tt = 1; tt < T_SZ; ++tt) {       // first-occurrence argmax
            float v = tm1[tt];
            if (v > best) { best = v; bi = tt; }
        }
        idx_tm1[b] = bi;
    }
}

// ---------------------------------------------------------------------------
// Init: t=0 columns of keys/prs/hs, tm_modified row 0 (= tm1) and [1][0]
// ---------------------------------------------------------------------------
__global__ __launch_bounds__(256) void rnn_init(
    const float* __restrict__ inputs, float* __restrict__ out)
{
    int tid = blockIdx.x * 256 + threadIdx.x;     // grid covers B*T = 262144
    int b = tid >> 10, t = tid & (T_SZ - 1);
    out[PT + b * 2 * T_SZ + t] = inputs[b * C_SZ * T_SZ + 6 * T_SZ + t];
    if (t == 0)
        out[PT + b * 2 * T_SZ + T_SZ] = inputs[b * C_SZ * T_SZ + 7 * T_SZ];
    if (tid < B_SZ * H_SZ) out[PH + tid * T_SZ] = 0.f;   // hs[:,:,0]
    if (tid < B_SZ * 2)    out[PK + tid * T_SZ] = 0.f;   // keys[:,:,0]
    if (tid < B_SZ * 4)    out[PP + tid * T_SZ] = 0.f;   // prs[:,:,0]
}

// ---------------------------------------------------------------------------
// Main persistent scan kernel: 16 blocks x 256 threads (8 waves).
// Block handles 16 batch rows; each wave owns 4 N-tiles of 16 columns.
// K-range [0,KC) of w_rec is pinned in LDS (async-filled once); the rest
// streams from L2 each step.
// ---------------------------------------------------------------------------
__global__ __launch_bounds__(256) void rnn_main(
    const float* __restrict__ inputs, const float* __restrict__ w_in,
    const float* __restrict__ b_in,   const float* __restrict__ b_rec,
    const float* __restrict__ w_key,  const float* __restrict__ b_key,
    const float* __restrict__ z,      const unsigned short* __restrict__ wrec_bf,
    const int* __restrict__ idx_tm1,  float* __restrict__ out)
{
    __shared__ __attribute__((aligned(32))) unsigned short Wc[H_SZ][KC];      // 256 KB W-cache
    __shared__ __attribute__((aligned(32))) unsigned short Abuf[2][16][H_SZ]; // 32 KB, dbl-buffered h
    __shared__ float inp_s[16][6];
    __shared__ float kpart[2][16][8];    // [key col][row m][wave]
    __shared__ float key_pre[16];
    __shared__ int   side_pre[16];
    __shared__ float off_s[16];

    const int tid = threadIdx.x, lane = tid & 31, wave = tid >> 5;
    const int hi = lane >> 4, lp = lane & 15;
    const int blockBase = blockIdx.x * 16;

    // ---- One-time: async-fill the LDS W-cache (rows n, k < KC) ----
    {
        unsigned ldsBase = (unsigned)(uintptr_t)&Wc[0][0];   // low 32 bits = LDS offset
        #pragma unroll 4
        for (int i = 0; i < (H_SZ * KC) / (8 * 256); ++i) {  // 64 x 16B chunks/thread
            int chunk = tid * 64 + i;                        // 0 .. 16383
            int n  = chunk >> 5;                             // 32 chunks per row
            int kk = (chunk & 31) * 8;
            unsigned lds_addr = ldsBase + (unsigned)((n * KC + kk) * 2);
            unsigned long long ga =
                (unsigned long long)(uintptr_t)(wrec_bf + n * H_SZ + kk);
            asm volatile("global_load_async_to_lds_b128 %0, %1, off"
                         :: "v"(lds_addr), "v"(ga) : "memory");
        }
        unsigned short* ab = &Abuf[0][0][0];
        for (int i = tid; i < 16 * H_SZ; i += 256) ab[i] = 0;    // h0 = 0
        if (tid < 16) { key_pre[tid] = 1.f; side_pre[tid] = 1; off_s[tid] = 0.f; }
        asm volatile("s_wait_asynccnt 0x0" ::: "memory");
    }
    __syncthreads();

    // Per-lane loop-invariant constants (n is fixed per lane per tile)
    int   nlane[4], brow[4], hb[4];
    float bias[4], wk0r[4], wk1r[4], winr[4][6];
    #pragma unroll
    for (int nt = 0; nt < 4; ++nt) {
        int n = (wave * 4 + nt) * 16 + lp;
        nlane[nt] = n;
        brow[nt]  = n * H_SZ;
        bias[nt]  = b_in[n] + b_rec[n];
        wk0r[nt]  = w_key[n];
        wk1r[nt]  = w_key[H_SZ + n];
        #pragma unroll
        for (int j = 0; j < 6; ++j) winr[nt][j] = w_in[n * 6 + j];
        // hs flat base: PH + ((blockBase + 8*hi)*H + n)*T ; add v*H*T + t per store
        hb[nt] = PH + (blockBase + 8 * hi) * (H_SZ * T_SZ) + n * T_SZ;
    }

    int idxb = 0, bgl = 0;
    const float* inb = nullptr;
    if (tid < 16) {
        bgl  = blockBase + tid;
        inb  = inputs + bgl * C_SZ * T_SZ;
        idxb = idx_tm1[bgl];
    }

    #pragma unroll 1
    for (int t = 1; t < T_SZ; ++t) {
        const int cur = (t - 1) & 1, nxt = t & 1;

        // ---- Phase 1: per-row scalar logic (threads 0..15) ----
        if (tid < 16) {
            const int m = tid;
            float ap0 = inb[t],            ap1 = inb[T_SZ + t];
            float as0 = inb[2 * T_SZ + t], as1 = inb[3 * T_SZ + t];
            float as2 = inb[4 * T_SZ + t], as3 = inb[5 * T_SZ + t];
            float t1  = inb[6 * T_SZ + t], t2o = inb[7 * T_SZ + t];
            float zt  = z[(t - 1) * B_SZ + bgl];
            float kp  = key_pre[m], omk = 1.f - kp;
            int side_h  = (kp > 0.5f);
            float asum0 = as0 * kp + as2 * omk;
            float asum1 = as1 * kp + as3 * omk;
            int closer1 = (fabsf(ap0 - t1) <= fabsf(ap1 - t1));  // argmin==0, ties->first
            int done    = (t > idxb);
            int switched = (side_h != side_pre[m]) && done && (side_h != closer1);
            float off = off_s[m];
            float t2c = t2o + off;
            if (switched) off += zt * fabsf(WB_INCf * t2c);
            off_s[m] = off;
            float t2e = t2o + off;
            side_pre[m] = side_h;
            out[PP + (bgl * 4 + 0) * T_SZ + t] = lik(t1, ap0);
            out[PP + (bgl * 4 + 1) * T_SZ + t] = lik(t1, ap1);
            out[PP + (bgl * 4 + 2) * T_SZ + t] = lik(t2e, asum0);
            out[PP + (bgl * 4 + 3) * T_SZ + t] = lik(t2e, asum1);
            out[PT + (bgl * 2 + 1) * T_SZ + t] = t2e;
            inp_s[m][0] = ap0;  inp_s[m][1] = ap1;
            inp_s[m][2] = asum0; inp_s[m][3] = asum1;
            inp_s[m][4] = t1;   inp_s[m][5] = t2e;
        }
        __syncthreads();

        // ---- Phase 2: h_t = tanh(inp@w_in.T + b + h_pre@w_rec.T) via WMMA ----
        v8f acc[4];
        #pragma unroll
        for (int v = 0; v < 8; ++v) {                 // fold input projection + biases
            const int m = v + 8 * hi;
            float i0 = inp_s[m][0], i1 = inp_s[m][1], i2 = inp_s[m][2];
            float i3 = inp_s[m][3], i4 = inp_s[m][4], i5 = inp_s[m][5];
            #pragma unroll
            for (int nt = 0; nt < 4; ++nt) {
                acc[nt][v] = bias[nt]
                    + i0 * winr[nt][0] + i1 * winr[nt][1] + i2 * winr[nt][2]
                    + i3 * winr[nt][3] + i4 * winr[nt][4] + i5 * winr[nt][5];
            }
        }

        const unsigned short* Arow = &Abuf[cur][lp][0];
        const unsigned short* WcBase = &Wc[0][0];

        // K-half 2 first (kt 8..15): B streams from L2 -> loads fly early
        #pragma unroll 4
        for (int kt = 8; kt < 16; ++kt) {
            const int k0 = kt * 32;
            bf16x8 alo = *(const bf16x8*)(Arow + k0 + 16 * hi);
            bf16x8 ahi = *(const bf16x8*)(Arow + k0 + 16 * hi + 8);
            v16bf afrag = __builtin_shufflevector(alo, ahi,
                0, 1, 2, 3, 4, 5, 6, 7, 8, 9, 10, 11, 12, 13, 14, 15);
            #pragma unroll
            for (int nt = 0; nt < 4; ++nt) {
                v16bf bfrag = *(const v16bf*)(wrec_bf + brow[nt] + k0 + 16 * hi);
                acc[nt] = __builtin_amdgcn_wmma_f32_16x16x32_bf16(
                    false, afrag, false, bfrag, (short)0, acc[nt], false, false);
            }
        }
        // K-half 1 (kt 0..7): B from the pinned LDS W-cache
        #pragma unroll 4
        for (int kt = 0; kt < 8; ++kt) {
            const int k0 = kt * 32;
            bf16x8 alo = *(const bf16x8*)(Arow + k0 + 16 * hi);
            bf16x8 ahi = *(const bf16x8*)(Arow + k0 + 16 * hi + 8);
            v16bf afrag = __builtin_shufflevector(alo, ahi,
                0, 1, 2, 3, 4, 5, 6, 7, 8, 9, 10, 11, 12, 13, 14, 15);
            #pragma unroll
            for (int nt = 0; nt < 4; ++nt) {
                v16bf bfrag = *(const v16bf*)(WcBase + nlane[nt] * KC + k0 + 16 * hi);
                acc[nt] = __builtin_amdgcn_wmma_f32_16x16x32_bf16(
                    false, afrag, false, bfrag, (short)0, acc[nt], false, false);
            }
        }

        // ---- Epilogue: tanh, hs store, bf16 write-back, key partials ----
        float kp0v[8], kp1v[8];
        #pragma unroll
        for (int v = 0; v < 8; ++v) { kp0v[v] = 0.f; kp1v[v] = 0.f; }
        #pragma unroll
        for (int nt = 0; nt < 4; ++nt) {
            const int n = nlane[nt];
            #pragma unroll
            for (int v = 0; v < 8; ++v) {
                float h = fast_tanh(acc[nt][v]);                // D elem (m=v+8*hi, n)
                out[hb[nt] + v * (H_SZ * T_SZ) + t] = h;        // hs[b, n, t]
                Abuf[nxt][v + 8 * hi][n] = f32_to_bf16(h);      // next step's A
                kp0v[v] += h * wk0r[nt];
                kp1v[v] += h * wk1r[nt];
            }
        }
        // deterministic in-wave butterfly reduce over the 16-lane group, then LDS
        #pragma unroll
        for (int v = 0; v < 8; ++v) {
            float s0 = kp0v[v], s1 = kp1v[v];
            #pragma unroll
            for (int msk = 8; msk >= 1; msk >>= 1) {
                s0 += __shfl_xor(s0, msk, 32);
                s1 += __shfl_xor(s1, msk, 32);
            }
            if (lp == 0) {
                kpart[0][v + 8 * hi][wave] = s0;
                kpart[1][v + 8 * hi][wave] = s1;
            }
        }
        __syncthreads();

        // ---- Phase 3: key head (threads 0..31), fixed-order sum -> sigmoid ----
        if (tid < 32) {
            const int c = tid >> 4, m = tid & 15;    // c==0 handled by thread m itself
            float s = b_key[c];
            #pragma unroll
            for (int w = 0; w < 8; ++w) s += kpart[c][m][w];
            float k = 1.f / (1.f + __expf(-s));
            out[PK + ((blockBase + m) * 2 + c) * T_SZ + t] = k;
            if (c == 0) key_pre[m] = k;              // same thread reads it next step
        }
        // no barrier needed here: carries are same-thread; kpart rewritten only
        // after the next iteration's first barrier.
    }
}

// ---------------------------------------------------------------------------
extern "C" void kernel_launch(void* const* d_in, const int* in_sizes, int n_in,
                              void* d_out, int out_size, void* d_ws, size_t ws_size,
                              hipStream_t stream)
{
    (void)in_sizes; (void)n_in; (void)out_size; (void)ws_size;
    const float* inputs = (const float*)d_in[0];
    const float* w_in   = (const float*)d_in[1];
    const float* b_in   = (const float*)d_in[2];
    const float* w_rec  = (const float*)d_in[3];
    const float* b_rec  = (const float*)d_in[4];
    const float* w_key  = (const float*)d_in[5];
    const float* b_key  = (const float*)d_in[6];
    const float* z      = (const float*)d_in[7];
    float* out = (float*)d_out;

    unsigned short* wrec_bf = (unsigned short*)d_ws;
    int* idx = (int*)((char*)d_ws + (size_t)H_SZ * H_SZ * sizeof(unsigned short));

    rnn_prep<<<64, 256, 0, stream>>>(w_rec, wrec_bf, inputs, idx);
    rnn_init<<<1024, 256, 0, stream>>>(inputs, out);
    rnn_main<<<16, 256, 0, stream>>>(inputs, w_in, b_in, b_rec, w_key, b_key,
                                     z, wrec_bf, idx, out);
}
